// SelfAttention_28260884807977
// MI455X (gfx1250) — matrix operand order repaired and checked by reading, not compiled, not run on hardware
//
#include <hip/hip_runtime.h>
#include <hip/hip_bf16.h>

// ---------------------------------------------------------------------------
// Complex self-attention (time + freq) for MI455X / gfx1250, wave32 + WMMA.
// B=2, C=64, F=256, T=256.  All GEMMs via v_wmma_f32_16x16x32_bf16.
// Global->LDS staging of pure-copy tiles uses GLOBAL_LOAD_ASYNC_TO_LDS_B128
// (ASYNCcnt) — builtin confirmed present on this toolchain.
// ---------------------------------------------------------------------------

typedef __attribute__((ext_vector_type(16))) __bf16 v16bf;
typedef __attribute__((ext_vector_type(8)))  float  v8f;
typedef int v4i __attribute__((vector_size(16)));    // matches builtin param

#define BATCH 2
#define CCH   64
#define FF    256
#define TT    256
#define FT    65536            // F*T
#define NELEM 16777216         // B*2*C*F*T (one branch buffer)
#define BFTCNT 131072.0f       // B*F*T  (BN reduction count)

// ---- bf16 bit helpers (storage = unsigned short) --------------------------
static __device__ __forceinline__ unsigned short f2bf(float f) {
    unsigned u = __float_as_uint(f);
    unsigned r = (u + 0x7fffu + ((u >> 16) & 1u)) >> 16;   // RNE
    return (unsigned short)r;
}
static __device__ __forceinline__ float bf2f(unsigned short s) {
    return __uint_as_float(((unsigned)s) << 16);
}

// ---- async global->LDS copy (16B per lane), CDNA5 path with fallback ------
#if defined(__HIP_DEVICE_COMPILE__) && defined(__has_builtin)
#if __has_builtin(__builtin_amdgcn_global_load_async_to_lds_b128)
#define HAVE_ASYNC_LDS 1
#endif
#endif

static __device__ __forceinline__ void async_cp16(const void* g, void* l) {
#ifdef HAVE_ASYNC_LDS
    __builtin_amdgcn_global_load_async_to_lds_b128(
        (__attribute__((address_space(1))) v4i*)(unsigned long long)g,
        (__attribute__((address_space(3))) v4i*)(unsigned)(unsigned long long)l,
        0, 0);
#else
    *reinterpret_cast<uint4*>(l) = *reinterpret_cast<const uint4*>(g);
#endif
}
static __device__ __forceinline__ void async_wait() {
#ifdef HAVE_ASYNC_LDS
#if __has_builtin(__builtin_amdgcn_s_wait_asynccnt)
    __builtin_amdgcn_s_wait_asynccnt(0);
#else
    asm volatile("s_wait_asynccnt 0" ::: "memory");
#endif
#endif
}

// ---- WMMA fragment loaders (documented wave32 layouts) --------------------
// A (16x32 bf16): lane l holds row M=l&15; lanes<16 get K {0..7,16..23},
// lanes>=16 get K {8..15,24..31}  -> two 16B chunks per lane.
static __device__ __forceinline__ v16bf ldA(const unsigned short* rowK0, int lane) {
    union { v16bf v; uint4 q[2]; } u;
    const int off = (lane & 16) ? 8 : 0;
    u.q[0] = *reinterpret_cast<const uint4*>(rowK0 + off);
    u.q[1] = *reinterpret_cast<const uint4*>(rowK0 + off + 16);
    return u.v;
}
// B (32x16 bf16): lane holds column N=l&15; K order same as A (strided gather).
static __device__ __forceinline__ v16bf ldB16(const unsigned short* p, int stride, int lane) {
    union { v16bf v; unsigned short h[16]; } u;
    const int off = (lane & 16) ? 8 : 0;
    for (int s = 0; s < 8; ++s) {
        u.h[s]     = p[(off + s) * stride];
        u.h[s + 8] = p[(16 + off + s) * stride];
    }
    return u.v;
}
static __device__ __forceinline__ v8f wmma_bf16(v16bf a, v16bf b, v8f c) {
    return __builtin_amdgcn_wmma_f32_16x16x32_bf16(false, a, false, b,
                                                   short(0), c, false, false);
}

// ---------------------------------------------------------------------------
__global__ __launch_bounds__(256) void k_zero(float* p, int n) {
    int i = blockIdx.x * 256 + threadIdx.x;
    if (i < n) p[i] = 0.0f;
}

__global__ __launch_bounds__(256) void k_cvt_w(const float* W, unsigned short* Wbf, int n) {
    int i = blockIdx.x * 256 + threadIdx.x;
    if (i < n) Wbf[i] = f2bf(W[i]);
}

// ---------------------------------------------------------------------------
// Complex 1x1 conv: y[b,r,o,p] for one branch.  (bias cancels through BN.)
//   yr = W0*xr - W1*xi ; yi = W0*xi + W1*xr
// grid: (512 ft-tiles of 128, B) block 256 (8 waves)
// ---------------------------------------------------------------------------
__global__ __launch_bounds__(256) void k_conv(const float* __restrict__ x,
                                              const unsigned short* __restrict__ Wbf,
                                              float* __restrict__ y) {
    __shared__ unsigned short Ws[3][CCH][CCH];   // W0, W1, -W1   (24 KB)
    __shared__ unsigned short Xs[2][CCH][128];   // xr, xi tiles  (32 KB)

    const int tid = threadIdx.x;
    const int b   = blockIdx.y;
    const int p0  = blockIdx.x * 128;

    // stage weights (2*64*64 bf16)
    for (int idx = tid; idx < 8192; idx += 256) {
        int w01 = idx >> 12, o = (idx >> 6) & 63, c = idx & 63;
        unsigned short v = Wbf[idx];
        Ws[w01][o][c] = v;
        if (w01 == 1) Ws[2][o][c] = f2bf(-bf2f(v));
    }
    // stage x tile: 2 reim * 64 ch * 128 pos (fp32 -> bf16)
    {
        int r = tid >> 7, c = (tid >> 1) & 63, half = tid & 1;
        const float* src = x + ((size_t)((b * 2 + r) * CCH + c)) * FT + p0 + half * 64;
        for (int j = 0; j < 16; ++j) {
            float4 v = reinterpret_cast<const float4*>(src)[j];
            int o = half * 64 + j * 4;
            Xs[r][c][o + 0] = f2bf(v.x);
            Xs[r][c][o + 1] = f2bf(v.y);
            Xs[r][c][o + 2] = f2bf(v.z);
            Xs[r][c][o + 3] = f2bf(v.w);
        }
    }
    __syncthreads();

    const int wave = tid >> 5, lane = tid & 31;
    const int pl = wave * 16;                        // ntile position offset
    for (int mtile = 0; mtile < 4; ++mtile) {
        v8f accR = {}, accI = {};
        for (int kc = 0; kc < 2; ++kc) {
            const int row = mtile * 16 + (lane & 15);
            v16bf a0 = ldA(&Ws[0][row][kc * 32], lane);
            v16bf a1 = ldA(&Ws[1][row][kc * 32], lane);
            v16bf am = ldA(&Ws[2][row][kc * 32], lane);
            v16bf br = ldB16(&Xs[0][kc * 32][pl + (lane & 15)], 128, lane);
            v16bf bi = ldB16(&Xs[1][kc * 32][pl + (lane & 15)], 128, lane);
            accR = wmma_bf16(a0, br, accR);
            accR = wmma_bf16(am, bi, accR);          // -W1 * xi
            accI = wmma_bf16(a0, bi, accI);
            accI = wmma_bf16(a1, br, accI);
        }
        // C/D layout: VGPR j -> M = j + 8*(lane>=16), N = lane&15
        const int o = mtile * 16 + ((lane >> 4) << 3);
        const int p = p0 + pl + (lane & 15);
        float* yR = y + ((size_t)((b * 2 + 0) * CCH + o)) * FT + p;
        float* yI = y + ((size_t)((b * 2 + 1) * CCH + o)) * FT + p;
        for (int j = 0; j < 8; ++j) {
            yR[(size_t)j * FT] = accR[j];
            yI[(size_t)j * FT] = accI[j];
        }
    }
}

// ---------------------------------------------------------------------------
// Per-channel sum / sumsq of y (one branch).  grid (256 rows, 8 chunks)
// statsBr layout: [(r*64+o)*2 + {sum,sumsq}]
// ---------------------------------------------------------------------------
__global__ __launch_bounds__(256) void k_stats(const float* __restrict__ y,
                                               float* __restrict__ statsBr) {
    __shared__ float rs[256], rq[256];
    const int tid = threadIdx.x;
    const int row = blockIdx.x;                   // (b*2+r)*64+o
    const size_t base = (size_t)row * FT + blockIdx.y * 8192;
    float s = 0.f, q = 0.f;
    for (int i = tid; i < 8192; i += 256) { float v = y[base + i]; s += v; q += v * v; }
    rs[tid] = s; rq[tid] = q; __syncthreads();
    for (int off = 128; off > 0; off >>= 1) {
        if (tid < off) { rs[tid] += rs[tid + off]; rq[tid] += rq[tid + off]; }
        __syncthreads();
    }
    if (tid == 0) {
        int ro = row & 127;                       // fold b away
        atomicAdd(statsBr + ro * 2 + 0, rs[0]);
        atomicAdd(statsBr + ro * 2 + 1, rq[0]);
    }
}

__global__ __launch_bounds__(128) void k_bnfinal(const float* statsBr,
                                                 const float* g, const float* be,
                                                 float* scshBr) {
    int t = threadIdx.x;                          // r*64+o
    if (t < 128) {
        float mu  = statsBr[t * 2 + 0] / BFTCNT;
        float var = statsBr[t * 2 + 1] / BFTCNT - mu * mu;
        float sc  = g[t] * rsqrtf(var + 1e-5f);
        scshBr[t * 2 + 0] = sc;
        scshBr[t * 2 + 1] = be[t] - mu * sc;
    }
}

static __device__ __forceinline__ float lrelu(float a) { return a > 0.f ? a : 0.01f * a; }

// ---------------------------------------------------------------------------
// Normalize + leaky relu, write bf16 TRANSPOSED (b,r,t, c*256+f) (time branches)
// grid: (16 ft-tiles(64x64), 64 channels, 4 b*2+r)
// ---------------------------------------------------------------------------
__global__ __launch_bounds__(256) void k_norm_time(const float* __restrict__ y,
                                                   const float* __restrict__ scshBr,
                                                   unsigned short* __restrict__ dst) {
    __shared__ unsigned short tile[64][66];
    const int tid = threadIdx.x;
    const int b2r = blockIdx.z, c = blockIdx.y;
    const int f0 = (blockIdx.x >> 2) * 64, t0 = (blockIdx.x & 3) * 64;
    const int r = b2r & 1;
    const float sc = scshBr[(r * 64 + c) * 2 + 0];
    const float sh = scshBr[(r * 64 + c) * 2 + 1];

    {
        int fl = tid >> 2, seg = tid & 3;
        const float* src = y + ((size_t)(b2r * CCH + c)) * FT + (f0 + fl) * TT + t0 + seg * 16;
        for (int j = 0; j < 4; ++j) {
            float4 v = reinterpret_cast<const float4*>(src)[j];
            int o = seg * 16 + j * 4;
            tile[fl][o + 0] = f2bf(lrelu(sc * v.x + sh));
            tile[fl][o + 1] = f2bf(lrelu(sc * v.y + sh));
            tile[fl][o + 2] = f2bf(lrelu(sc * v.z + sh));
            tile[fl][o + 3] = f2bf(lrelu(sc * v.w + sh));
        }
    }
    __syncthreads();
    {
        int tl = tid >> 2, seg = tid & 3;
        union { unsigned short h[16]; uint4 q[2]; } u;
        for (int j = 0; j < 16; ++j) u.h[j] = tile[seg * 16 + j][tl];
        unsigned short* d = dst + ((size_t)(b2r * TT + t0 + tl)) * 16384 + c * 256 + f0 + seg * 16;
        reinterpret_cast<uint4*>(d)[0] = u.q[0];
        reinterpret_cast<uint4*>(d)[1] = u.q[1];
    }
}

// Normalize + leaky relu, natural layout (freq branches). grid 16384
__global__ __launch_bounds__(256) void k_norm_freq(const float* __restrict__ y,
                                                   const float* __restrict__ scshBr,
                                                   unsigned short* __restrict__ dst) {
    size_t i = ((size_t)blockIdx.x * 256 + threadIdx.x) * 4;
    int ro = (int)((i >> 16) & 127);              // r*64+o
    float sc = scshBr[ro * 2 + 0], sh = scshBr[ro * 2 + 1];
    float4 v = *reinterpret_cast<const float4*>(y + i);
    union { unsigned short h[4]; uint2 u; } o;
    o.h[0] = f2bf(lrelu(sc * v.x + sh));
    o.h[1] = f2bf(lrelu(sc * v.y + sh));
    o.h[2] = f2bf(lrelu(sc * v.z + sh));
    o.h[3] = f2bf(lrelu(sc * v.w + sh));
    *reinterpret_cast<uint2*>(dst + i) = o.u;
}

// ---------------------------------------------------------------------------
// Attention scores + column softmax.
// element(row n, d) = base + b*bS + r*rS + n*rowS + (d>>8)*chS + (d&255)
// grid: (8 m-strips of 32, B) block 256.
// ---------------------------------------------------------------------------
__global__ __launch_bounds__(256) void k_scores(const unsigned short* __restrict__ qb,
                                                const unsigned short* __restrict__ kb,
                                                float* __restrict__ aOut,
                                                long long bS, long long rS,
                                                int rowS, int chS) {
    __shared__ unsigned short Qs[2][256][32];     // 32 KB
    __shared__ unsigned short Ks[2][32][40];      // transposed [k][m], 5 KB
    __shared__ float ampL[256][32];               // 32 KB

    const int tid = threadIdx.x;
    const int b = blockIdx.y;
    const int m0 = blockIdx.x * 32;
    const int wave = tid >> 5, lane = tid & 31;
    const int ntA = wave * 2;

    v8f accS[2][2], accIR[2][2], accRI[2][2];
    v8f z = {};
    for (int i = 0; i < 2; ++i)
        for (int m = 0; m < 2; ++m) { accS[i][m] = z; accIR[i][m] = z; accRI[i][m] = z; }

    for (int kc = 0; kc < 512; ++kc) {
        const int hi = (kc * 32) >> 8, lo = (kc * 32) & 255;
        // stage Q rows (both reim): 2048 16B segments, async copy to LDS
        for (int idx = tid; idx < 2048; idx += 256) {
            int seg = idx & 3, n = (idx >> 2) & 255, rr = idx >> 10;
            const unsigned short* src = qb + b * bS + rr * rS + (long long)n * rowS
                                           + (long long)hi * chS + lo + seg * 8;
            async_cp16(src, &Qs[rr][n][seg * 8]);
        }
        // stage K rows transposed: 64 jobs (register transpose -> ds scatter)
        if (tid < 64) {
            int rr = tid >> 5, m = tid & 31;
            const unsigned short* src = kb + b * bS + rr * rS + (long long)(m0 + m) * rowS
                                           + (long long)hi * chS + lo;
            union { uint4 q[4]; unsigned short h[32]; } u;
            for (int j = 0; j < 4; ++j) u.q[j] = reinterpret_cast<const uint4*>(src)[j];
            for (int k = 0; k < 32; ++k) Ks[rr][k][m] = u.h[k];
        }
        async_wait();
        __syncthreads();

        v16bf aq[2][2], bk[2][2];
        for (int rr = 0; rr < 2; ++rr) {
            for (int i = 0; i < 2; ++i)
                aq[rr][i] = ldA(&Qs[rr][(ntA + i) * 16 + (lane & 15)][0], lane);
            for (int mt = 0; mt < 2; ++mt)
                bk[rr][mt] = ldB16(&Ks[rr][0][mt * 16 + (lane & 15)], 40, lane);
        }
        for (int i = 0; i < 2; ++i)
            for (int mt = 0; mt < 2; ++mt) {
                accS[i][mt]  = wmma_bf16(aq[0][i], bk[0][mt], accS[i][mt]);  // qr.kr
                accS[i][mt]  = wmma_bf16(aq[1][i], bk[1][mt], accS[i][mt]);  // +qi.ki
                accIR[i][mt] = wmma_bf16(aq[1][i], bk[0][mt], accIR[i][mt]); // qi.kr
                accRI[i][mt] = wmma_bf16(aq[0][i], bk[1][mt], accRI[i][mt]); // qr.ki
            }
        __syncthreads();
    }
    // amplitude -> LDS
    for (int i = 0; i < 2; ++i)
        for (int mt = 0; mt < 2; ++mt)
            for (int j = 0; j < 8; ++j) {
                int n = (ntA + i) * 16 + j + ((lane >> 4) << 3);
                int m = mt * 16 + (lane & 15);
                float sr = accS[i][mt][j];
                float si = accIR[i][mt][j] - accRI[i][mt][j];
                ampL[n][m] = sqrtf(sr * sr + si * si);
            }
    __syncthreads();
    // softmax over n (axis=1), per column m
    if (tid < 32) {
        int m = tid;
        float mx = -3.4e38f;
        for (int n = 0; n < 256; ++n) mx = fmaxf(mx, ampL[n][m]);
        float sum = 0.f;
        for (int n = 0; n < 256; ++n) { float e = __expf(ampL[n][m] - mx); ampL[n][m] = e; sum += e; }
        float inv = 1.0f / sum;
        for (int n = 0; n < 256; ++n) ampL[n][m] *= inv;
    }
    __syncthreads();
    for (int idx = tid; idx < 8192; idx += 256) {
        int n = idx >> 5, m = idx & 31;
        aOut[(size_t)b * 65536 + n * 256 + m0 + m] = ampL[n][m];
    }
}

// ---------------------------------------------------------------------------
// out[n,d] = sum_m a[n,m] * v[m,d]  for both reim planes of v.
// v element(m,d) addressed like k_scores.  Output goes straight to d_out:
//   oidx = ((b*2+r)*64 + (d>>8))*65536 + (d&255)*dLowS + n*nS    (+= if add)
// grid: (256 d-blocks of 64, 2 n-halves of 128, B) block 256.
// ---------------------------------------------------------------------------
__global__ __launch_bounds__(256) void k_attnout(const float* __restrict__ aBuf,
                                                 const unsigned short* __restrict__ vb,
                                                 float* __restrict__ out,
                                                 long long bS, long long rS,
                                                 int rowS, int chS,
                                                 int dLowS, int nS, int addFlag) {
    __shared__ unsigned short As[128][32];        // 8 KB (a in bf16)
    __shared__ unsigned short Vs[2][32][64];      // 8 KB

    const int tid = threadIdx.x;
    const int b = blockIdx.z;
    const int n0 = blockIdx.y * 128;
    const int dblk = blockIdx.x;
    const int hi = dblk >> 2, dlo0 = (dblk & 3) * 64;
    const int wave = tid >> 5, lane = tid & 31;

    v8f acc[2][4];
    v8f z = {};
    for (int r = 0; r < 2; ++r) for (int d = 0; d < 4; ++d) acc[r][d] = z;

    for (int mc = 0; mc < 8; ++mc) {
        const int mb = mc * 32;
        // stage V: 2 reim x 32 rows x 64 d -> 512 16B segments, async copy
        for (int idx = tid; idx < 512; idx += 256) {
            int seg = idx & 7, m = (idx >> 3) & 31, rr = idx >> 8;
            const unsigned short* src = vb + b * bS + rr * rS + (long long)(mb + m) * rowS
                                           + (long long)hi * chS + dlo0 + seg * 8;
            async_cp16(src, &Vs[rr][m][seg * 8]);
        }
        // stage A: 128 rows x 32 fp32 -> bf16 (conversion: synchronous)
        {
            int n = tid >> 1, half = tid & 1;
            const float* src = aBuf + (size_t)b * 65536 + (n0 + n) * 256 + mb + half * 16;
            union { unsigned short h[16]; uint4 q[2]; } u;
            for (int j = 0; j < 16; ++j) u.h[j] = f2bf(src[j]);
            uint4* d = reinterpret_cast<uint4*>(&As[n][half * 16]);
            d[0] = u.q[0]; d[1] = u.q[1];
        }
        async_wait();
        __syncthreads();

        v16bf af = ldA(&As[wave * 16 + (lane & 15)][0], lane);
        for (int rr = 0; rr < 2; ++rr)
            for (int dt = 0; dt < 4; ++dt) {
                v16bf bv = ldB16(&Vs[rr][0][dt * 16 + (lane & 15)], 64, lane);
                acc[rr][dt] = wmma_bf16(af, bv, acc[rr][dt]);
            }
        __syncthreads();
    }
    // epilogue
    for (int rr = 0; rr < 2; ++rr)
        for (int dt = 0; dt < 4; ++dt)
            for (int j = 0; j < 8; ++j) {
                int n = n0 + wave * 16 + j + ((lane >> 4) << 3);
                int d = dblk * 64 + dt * 16 + (lane & 15);
                size_t oidx = ((size_t)((b * 2 + rr) * CCH + (d >> 8))) * FT
                            + (size_t)(d & 255) * dLowS + (size_t)n * nS;
                float v = acc[rr][dt][j];
                if (addFlag) out[oidx] += v; else out[oidx] = v;
            }
}

// ---------------------------------------------------------------------------
extern "C" void kernel_launch(void* const* d_in, const int* in_sizes, int n_in,
                              void* d_out, int out_size, void* d_ws, size_t ws_size,
                              hipStream_t stream) {
    const float* Pf = (const float*)d_in[0];
    const float* Qf = (const float*)d_in[1];
    const float* Wf = (const float*)d_in[2];
    // d_in[3] (bias) cancels exactly through BatchNorm -> unused.
    const float* g  = (const float*)d_in[4];
    const float* be = (const float*)d_in[5];
    float* out = (float*)d_out;

    char* ws = (char*)d_ws;
    const size_t OFF_WBF   = 0;               //  98304 B
    const size_t OFF_STATS = 98304;           //   6144 B
    const size_t OFF_SCSH  = 104448;          //   6144 B
    const size_t OFF_Y     = 110592;          //  64 MiB (shared across branches)
    const size_t OFF_BB    = OFF_Y + (size_t)NELEM * 4;
    const size_t OFF_AT    = OFF_BB + (size_t)6 * NELEM * 2;
    const size_t OFF_AF    = OFF_AT + 524288;

    unsigned short* Wbf  = (unsigned short*)(ws + OFF_WBF);
    float* stats         = (float*)(ws + OFF_STATS);
    float* scsh          = (float*)(ws + OFF_SCSH);
    float* ybuf          = (float*)(ws + OFF_Y);
    unsigned short* bbs  = (unsigned short*)(ws + OFF_BB);
    float* aT            = (float*)(ws + OFF_AT);
    float* aF            = (float*)(ws + OFF_AF);

    k_zero<<<6, 256, 0, stream>>>(stats, 1536);
    k_cvt_w<<<192, 256, 0, stream>>>(Wf, Wbf, 49152);

    for (int br = 0; br < 6; ++br) {
        const float* xsrc = (br == 0 || br == 3) ? Qf : Pf;
        unsigned short* bb = bbs + (size_t)br * NELEM;
        k_conv<<<dim3(512, BATCH), 256, 0, stream>>>(xsrc, Wbf + br * 8192, ybuf);
        k_stats<<<dim3(256, 8), 256, 0, stream>>>(ybuf, stats + br * 256);
        k_bnfinal<<<1, 128, 0, stream>>>(stats + br * 256, g + br * 128, be + br * 128,
                                         scsh + br * 256);
        if (br < 3)
            k_norm_time<<<dim3(16, 64, 4), 256, 0, stream>>>(ybuf, scsh + br * 256, bb);
        else
            k_norm_freq<<<16384, 256, 0, stream>>>(ybuf, scsh + br * 256, bb);
    }

    // time attention: layout (b,r,t,cf) -> rowS=16384, chS=256
    k_scores<<<dim3(8, BATCH), 256, 0, stream>>>(bbs + 0 * (size_t)NELEM,
                                                 bbs + 1 * (size_t)NELEM, aT,
                                                 8388608LL, 4194304LL, 16384, 256);
    // freq attention: natural (b,r,c,f,t) -> rowS(n=f)=256, chS(c)=65536
    k_scores<<<dim3(8, BATCH), 256, 0, stream>>>(bbs + 3 * (size_t)NELEM,
                                                 bbs + 4 * (size_t)NELEM, aF,
                                                 8388608LL, 4194304LL, 256, 65536);

    // ot written directly (=), of added (+=): d_out = ot + of
    k_attnout<<<dim3(256, 2, BATCH), 256, 0, stream>>>(aT, bbs + 2 * (size_t)NELEM, out,
                                                       8388608LL, 4194304LL, 16384, 256,
                                                       /*dLowS*/256, /*nS*/1, 0);
    k_attnout<<<dim3(256, 2, BATCH), 256, 0, stream>>>(aF, bbs + 5 * (size_t)NELEM, out,
                                                       8388608LL, 4194304LL, 256, 65536,
                                                       /*dLowS*/1, /*nS*/256, 1);
}
